// AttentionMemoryEntry_19662360281800
// MI455X (gfx1250) — matrix operand order repaired and checked by reading, not compiled, not run on hardware
//
#include <hip/hip_runtime.h>

// ---------------------------------------------------------------------------
// Types for CDNA5 WMMA (gfx1250, wave32)
// ---------------------------------------------------------------------------
typedef __bf16 bhalf;
typedef bhalf v16bf __attribute__((ext_vector_type(16)));
typedef float v8f   __attribute__((ext_vector_type(8)));

union Frag {
  uint4 q[2];           // 2 x 16B contiguous loads
  v16bf v;              // WMMA operand (16 bf16 = 8 VGPRs)
};

__device__ __forceinline__ unsigned short f32_to_bf16(float f) {
  unsigned int u = __float_as_uint(f);
  u += 0x7FFFu + ((u >> 16) & 1u);      // round-to-nearest-even
  return (unsigned short)(u >> 16);
}
__device__ __forceinline__ float bf16_to_f32(unsigned int h) {
  return __uint_as_float(h << 16);
}

// ---------------------------------------------------------------------------
// Elementwise f32 -> bf16 convert (row-major preserved) — for activations
// ---------------------------------------------------------------------------
__global__ void cvt_bf16_kernel(const float* __restrict__ in,
                                unsigned short* __restrict__ out, int n) {
  int i = blockIdx.x * blockDim.x + threadIdx.x;
  int stride = gridDim.x * blockDim.x;
  for (; i < n; i += stride) out[i] = f32_to_bf16(in[i]);
}

// ---------------------------------------------------------------------------
// Fused transpose + f32->bf16 convert: in[R,C] f32 -> out[C,R] bf16.
// 32x32 LDS tile keeps both global sides coalesced.
// ---------------------------------------------------------------------------
__global__ void cvt_transpose_bf16_kernel(const float* __restrict__ in,
                                          unsigned short* __restrict__ out,
                                          int R, int C) {
  __shared__ float tile[32][33];
  const int bx = blockIdx.x * 32;   // along C
  const int by = blockIdx.y * 32;   // along R
  const int tx = threadIdx.x;       // 0..31
  const int ty = threadIdx.y;       // 0..7
#pragma unroll
  for (int i = 0; i < 4; ++i) {
    int r = ty + i * 8;
    tile[r][tx] = in[(size_t)(by + r) * C + bx + tx];
  }
  __syncthreads();
#pragma unroll
  for (int i = 0; i < 4; ++i) {
    int r = ty + i * 8;             // output row chunk (along C)
    out[(size_t)(bx + r) * R + by + tx] = f32_to_bf16(tile[tx][r]);
  }
}

// ---------------------------------------------------------------------------
// argmax over last dim (C=65) minus 1  -> samples[token]
// ---------------------------------------------------------------------------
__global__ void argmax_kernel(const float* __restrict__ attn,
                              int* __restrict__ samples, int T, int C) {
  int t = blockIdx.x * blockDim.x + threadIdx.x;
  if (t >= T) return;
  const float* row = attn + (size_t)t * C;
  float best = row[0];
  int bi = 0;
  for (int j = 1; j < C; ++j) {
    float v = row[j];
    if (v > best) { best = v; bi = j; }   // strict > keeps first max (jnp.argmax)
  }
  samples[t] = bi - 1;
}

// ---------------------------------------------------------------------------
// LayerNorm: one block (128 threads) per row of D=512. Writes f32 + bf16.
// ---------------------------------------------------------------------------
__global__ void ln_kernel(const float* __restrict__ in,
                          const float* __restrict__ g,
                          const float* __restrict__ b,
                          float* __restrict__ outF,
                          unsigned short* __restrict__ outBF, int D) {
  __shared__ float red[128];
  __shared__ float s_mean, s_rstd;
  int tid = threadIdx.x;
  const float* row = in + (size_t)blockIdx.x * D;

  float s = 0.f;
  for (int c = tid; c < D; c += 128) s += row[c];
  red[tid] = s; __syncthreads();
  for (int st = 64; st > 0; st >>= 1) {
    if (tid < st) red[tid] += red[tid + st];
    __syncthreads();
  }
  if (tid == 0) s_mean = red[0] / (float)D;
  __syncthreads();
  float m = s_mean;
  __syncthreads();

  float sq = 0.f;
  for (int c = tid; c < D; c += 128) { float d = row[c] - m; sq += d * d; }
  red[tid] = sq; __syncthreads();
  for (int st = 64; st > 0; st >>= 1) {
    if (tid < st) red[tid] += red[tid + st];
    __syncthreads();
  }
  if (tid == 0) s_rstd = rsqrtf(red[0] / (float)D + 1e-5f);
  __syncthreads();
  float rs = s_rstd;

  for (int c = tid; c < D; c += 128) {
    float y = (row[c] - m) * rs * g[c] + b[c];
    size_t o = (size_t)blockIdx.x * D + c;
    outF[o] = y;
    if (outBF) outBF[o] = f32_to_bf16(y);
  }
}

// ---------------------------------------------------------------------------
// LN(st2) then scatter: x2 = x + (valid ? LN(st2) : 0); writes f32 + bf16
// ---------------------------------------------------------------------------
__global__ void ln_scatter_kernel(const float* __restrict__ st2,
                                  const float* __restrict__ g,
                                  const float* __restrict__ b,
                                  const float* __restrict__ x,
                                  const int* __restrict__ samples,
                                  float* __restrict__ x2F,
                                  unsigned short* __restrict__ x2BF, int D) {
  __shared__ float red[128];
  __shared__ float s_mean, s_rstd;
  int tid = threadIdx.x;
  int r = blockIdx.x;
  const float* row = st2 + (size_t)r * D;
  float keep = (samples[r] >= 0) ? 1.f : 0.f;

  float s = 0.f;
  for (int c = tid; c < D; c += 128) s += row[c];
  red[tid] = s; __syncthreads();
  for (int st = 64; st > 0; st >>= 1) {
    if (tid < st) red[tid] += red[tid + st];
    __syncthreads();
  }
  if (tid == 0) s_mean = red[0] / (float)D;
  __syncthreads();
  float m = s_mean;
  __syncthreads();

  float sq = 0.f;
  for (int c = tid; c < D; c += 128) { float d = row[c] - m; sq += d * d; }
  red[tid] = sq; __syncthreads();
  for (int st = 64; st > 0; st >>= 1) {
    if (tid < st) red[tid] += red[tid + st];
    __syncthreads();
  }
  if (tid == 0) s_rstd = rsqrtf(red[0] / (float)D + 1e-5f);
  __syncthreads();
  float rs = s_rstd;

  for (int c = tid; c < D; c += 128) {
    size_t o = (size_t)r * D + c;
    float ln = (row[c] - m) * rs * g[c] + b[c];
    float y = x[o] + keep * ln;
    x2F[o] = y;
    x2BF[o] = f32_to_bf16(y);
  }
}

// ---------------------------------------------------------------------------
// bf16 GEMM, v_wmma_f32_16x16x32_bf16, no LDS:
//   C[M,N] = A[M,K] (bf16 row-major) @ BT[N,K]^T (bf16, pre-transposed)
//            + bias (+ residual) (relu?) -> f32 and/or bf16.
// Block = 256 threads = 8 waves; wave tile 32(M) x 64(N) = 2x4 WMMA subtiles;
// block tile 128x128. Every operand fragment is two contiguous b128 loads.
// ---------------------------------------------------------------------------
__global__ void gemm_bf16_wmma(const unsigned short* __restrict__ A,
                               const unsigned short* __restrict__ BT,
                               const float* __restrict__ bias,
                               const float* __restrict__ residual,
                               float* __restrict__ outF,
                               unsigned short* __restrict__ outBF,
                               int M, int N, int K, int relu) {
  const int tid  = threadIdx.x;
  const int wave = tid >> 5;
  const int lane = tid & 31;
  const int lr   = lane & 15;
  const int hi   = lane >> 4;                 // 0: lanes 0-15, 1: lanes 16-31

  const int m0 = blockIdx.y * 128 + (wave >> 1) * 32;   // 4 waves along M
  const int n0 = blockIdx.x * 128 + (wave & 1) * 64;    // 2 waves along N

  v8f acc[2][4];
#pragma unroll
  for (int mi = 0; mi < 2; ++mi)
#pragma unroll
    for (int ni = 0; ni < 4; ++ni)
#pragma unroll
      for (int e = 0; e < 8; ++e) acc[mi][ni][e] = 0.f;

  const int koffA = hi ? 8 : 0;    // A lanes 16-31 hold K 8..15 / 24..31
  const int koffB = hi ? 8 : 0;    // B lanes 16-31 hold K 8..15 / 24..31

  for (int k0 = 0; k0 < K; k0 += 32) {
    Frag a[2], b[4];
#pragma unroll
    for (int mi = 0; mi < 2; ++mi) {
      const unsigned short* ap =
          A + (size_t)(m0 + mi * 16 + lr) * K + k0 + koffA;
      a[mi].q[0] = *(const uint4*)(ap);        // K  +0..7   (local)
      a[mi].q[1] = *(const uint4*)(ap + 16);   // K +16..23  (local)
    }
#pragma unroll
    for (int ni = 0; ni < 4; ++ni) {
      const unsigned short* bp =
          BT + (size_t)(n0 + ni * 16 + lr) * K + k0 + koffB;
      b[ni].q[0] = *(const uint4*)(bp);        // K  +0..7   (local)
      b[ni].q[1] = *(const uint4*)(bp + 16);   // K +16..23  (local)
    }
#pragma unroll
    for (int mi = 0; mi < 2; ++mi)
#pragma unroll
      for (int ni = 0; ni < 4; ++ni)
        acc[mi][ni] = __builtin_amdgcn_wmma_f32_16x16x32_bf16(
            false, a[mi].v, false, b[ni].v, (short)0, acc[mi][ni],
            false, false);
  }

  // Epilogue. C layout: element e -> row (hi ? e+8 : e), col = lane%16
#pragma unroll
  for (int ni = 0; ni < 4; ++ni) {
    const int colg = n0 + ni * 16 + lr;
    const float bv = bias ? bias[colg] : 0.f;
#pragma unroll
    for (int mi = 0; mi < 2; ++mi) {
#pragma unroll
      for (int e = 0; e < 8; ++e) {
        const int rowg = m0 + mi * 16 + (hi ? e + 8 : e);
        float v = acc[mi][ni][e] + bv;
        const size_t o = (size_t)rowg * N + colg;
        if (residual) v += residual[o];
        if (relu) v = fmaxf(v, 0.f);
        if (outF)  outF[o]  = v;
        if (outBF) outBF[o] = f32_to_bf16(v);
      }
    }
  }
}

// ---------------------------------------------------------------------------
// Per-token attention over the selected memory entry.
// Block = 128 threads, one block per token. H=8, dh=64, l_mem=128, d=512.
// ---------------------------------------------------------------------------
__global__ void attention_kernel(const float* __restrict__ q,
                                 const unsigned short* __restrict__ Kp,
                                 const unsigned short* __restrict__ Vp,
                                 const unsigned char* __restrict__ kmask,
                                 const int* __restrict__ samples,
                                 unsigned short* __restrict__ ctx) {
  constexpr int D = 512, LM = 128, HH = 8, DH = 64;
  __shared__ float qs[D];
  __shared__ float sc[HH][LM];

  const int tok = blockIdx.x;
  const int tid = threadIdx.x;
  int sRaw = samples[tok];
  const int s = sRaw < 0 ? 0 : sRaw;

  const float* qrow = q + (size_t)tok * D;
  for (int c = tid; c < D; c += 128) qs[c] = qrow[c];
  __syncthreads();

  // scores: thread tid == key position l (uint4 = 8 bf16 per load)
  {
    const int l = tid;
    const unsigned short* krow = Kp + ((size_t)(s * LM + l)) * D;
    const bool mv = kmask[s * LM + l] != 0;
#pragma unroll
    for (int h = 0; h < HH; ++h) {
      const int base = h * DH;
      const uint4* k4 = (const uint4*)(krow + base);
      float acc = 0.f;
#pragma unroll
      for (int c = 0; c < 8; ++c) {
        uint4 p = k4[c];
        const float* qp = qs + base + c * 8;
        acc += qp[0] * bf16_to_f32(p.x & 0xffffu);
        acc += qp[1] * bf16_to_f32(p.x >> 16);
        acc += qp[2] * bf16_to_f32(p.y & 0xffffu);
        acc += qp[3] * bf16_to_f32(p.y >> 16);
        acc += qp[4] * bf16_to_f32(p.z & 0xffffu);
        acc += qp[5] * bf16_to_f32(p.z >> 16);
        acc += qp[6] * bf16_to_f32(p.w & 0xffffu);
        acc += qp[7] * bf16_to_f32(p.w >> 16);
      }
      sc[h][l] = mv ? acc * 0.125f : -1e9f;   // 1/sqrt(64)
    }
  }
  __syncthreads();

  // softmax over l for head h = tid/16; thread covers 4 output dims
  const int h = tid >> 4;
  const int dim0 = h * DH + (tid & 15) * 4;
  float mx = -3.4e38f;
  for (int j = 0; j < LM; ++j) mx = fmaxf(mx, sc[h][j]);
  float sum = 0.f;
  for (int j = 0; j < LM; ++j) sum += expf(sc[h][j] - mx);
  const float inv = 1.f / sum;

  float a0 = 0.f, a1 = 0.f, a2 = 0.f, a3 = 0.f;
  for (int j = 0; j < LM; ++j) {
    const float p = expf(sc[h][j] - mx) * inv;
    const uint2 pv = *(const uint2*)(Vp + ((size_t)(s * LM + j)) * D + dim0);
    a0 += p * bf16_to_f32(pv.x & 0xffffu);
    a1 += p * bf16_to_f32(pv.x >> 16);
    a2 += p * bf16_to_f32(pv.y & 0xffffu);
    a3 += p * bf16_to_f32(pv.y >> 16);
  }
  unsigned short* crow = ctx + (size_t)tok * D + dim0;
  crow[0] = f32_to_bf16(a0);
  crow[1] = f32_to_bf16(a1);
  crow[2] = f32_to_bf16(a2);
  crow[3] = f32_to_bf16(a3);
}

// ---------------------------------------------------------------------------
// Host launch
// ---------------------------------------------------------------------------
static inline void launch_gemm(const unsigned short* A, const unsigned short* BT,
                               const float* bias, const float* residual,
                               float* outF, unsigned short* outBF,
                               int M, int N, int K, int relu, hipStream_t s) {
  dim3 grid(N / 128, M / 128);
  gemm_bf16_wmma<<<grid, dim3(256), 0, s>>>(A, BT, bias, residual, outF, outBF,
                                            M, N, K, relu);
}

extern "C" void kernel_launch(void* const* d_in, const int* in_sizes, int n_in,
                              void* d_out, int out_size, void* d_ws, size_t ws_size,
                              hipStream_t stream) {
  constexpr int Bc = 4, T = 256, D = 512, FF = 2048, NMEM = 64, LMEM = 128;
  constexpr int TOK = Bc * T;           // 1024
  constexpr int MEMROWS = NMEM * LMEM;  // 8192

  const float* dec_output   = (const float*)d_in[0];
  // d_in[1] = tgt_mask (unused by reference math)
  const float* mem_attn_out = (const float*)d_in[2];
  const float* enc_out_mem  = (const float*)d_in[3];
  const float* tgt_emb_mem  = (const float*)d_in[4];
  const unsigned char* tgt_mask_mem = (const unsigned char*)d_in[5];
  const float* Wq = (const float*)d_in[6];  const float* bq = (const float*)d_in[7];
  const float* Wk = (const float*)d_in[8];  const float* bk = (const float*)d_in[9];
  const float* Wv = (const float*)d_in[10]; const float* bv = (const float*)d_in[11];
  const float* Wo = (const float*)d_in[12]; const float* bo = (const float*)d_in[13];
  const float* g0 = (const float*)d_in[14]; const float* be0 = (const float*)d_in[15];
  const float* g1 = (const float*)d_in[16]; const float* be1 = (const float*)d_in[17];
  const float* W1 = (const float*)d_in[18]; const float* bf1 = (const float*)d_in[19];
  const float* W2 = (const float*)d_in[20]; const float* bf2 = (const float*)d_in[21];
  const float* W3 = (const float*)d_in[22]; const float* bf3 = (const float*)d_in[23];
  const float* W4 = (const float*)d_in[24]; const float* bf4 = (const float*)d_in[25];
  float* out = (float*)d_out;

  // ---- workspace layout (256B aligned) ----
  char* wsp = (char*)d_ws;
  size_t off = 0;
  auto take = [&](size_t bytes) -> char* {
    char* p = wsp + off;
    off += (bytes + 255) & ~(size_t)255;
    return p;
  };
  int*            samples = (int*)take(TOK * sizeof(int));
  float*          x_f32   = (float*)take((size_t)TOK * D * 4);
  unsigned short* x_bf    = (unsigned short*)take((size_t)TOK * D * 2);
  float*          st_f32  = (float*)take((size_t)TOK * D * 4);
  unsigned short* st_bf   = (unsigned short*)take((size_t)TOK * D * 2);
  unsigned short* ctx_bf  = (unsigned short*)take((size_t)TOK * D * 2);
  // transposed bf16 weights: WT[N,K]
  unsigned short* WqT = (unsigned short*)take((size_t)D * D * 2);
  unsigned short* WkT = (unsigned short*)take((size_t)D * D * 2);
  unsigned short* WvT = (unsigned short*)take((size_t)D * D * 2);
  unsigned short* WoT = (unsigned short*)take((size_t)D * D * 2);
  unsigned short* W1T = (unsigned short*)take((size_t)D * FF * 2);
  unsigned short* W2T = (unsigned short*)take((size_t)D * FF * 2);
  unsigned short* W3T = (unsigned short*)take((size_t)D * FF * 2);
  unsigned short* W4T = (unsigned short*)take((size_t)D * FF * 2);
  // region A: enc_bf (dead after Kp GEMM) -> reused for FFN hidden activations
  char* regA = take((size_t)MEMROWS * D * 2);
  unsigned short* enc_bf = (unsigned short*)regA;
  unsigned short* h_bf   = (unsigned short*)regA;   // [TOK,FF] bf16 (4MB)
  unsigned short* h2_bf  = (unsigned short*)regA;   // reuses same (h dead by then)
  // region B: emb_bf (dead after Vp GEMM) -> reused for q / st2 / x2 / x2_bf
  char* regB = take((size_t)MEMROWS * D * 2);
  unsigned short* emb_bf = (unsigned short*)regB;
  float*          q_f32  = (float*)(regB + 0);      // 2MB
  float*          st2    = (float*)(regB + (size_t)2 * 1024 * 1024);
  float*          x2     = (float*)(regB + (size_t)4 * 1024 * 1024);
  unsigned short* x2_bf  = (unsigned short*)(regB + (size_t)6 * 1024 * 1024);
  unsigned short* Kp_bf  = (unsigned short*)take((size_t)MEMROWS * D * 2);
  unsigned short* Vp_bf  = (unsigned short*)take((size_t)MEMROWS * D * 2);
  (void)ws_size; (void)in_sizes; (void)n_in; (void)out_size;

  auto cvt = [&](const float* src, unsigned short* dst, int n) {
    int grid = (n + 255) / 256; if (grid > 2048) grid = 2048;
    cvt_bf16_kernel<<<grid, 256, 0, stream>>>(src, dst, n);
  };
  auto cvtT = [&](const float* src, unsigned short* dst, int R, int C) {
    dim3 grid(C / 32, R / 32);
    cvt_transpose_bf16_kernel<<<grid, dim3(32, 8), 0, stream>>>(src, dst, R, C);
  };

  // 1) samples = argmax(mem_attn_out, -1) - 1
  argmax_kernel<<<(TOK + 255) / 256, 256, 0, stream>>>(mem_attn_out, samples,
                                                       TOK, NMEM + 1);
  // 2) x = layernorm(dec_output, g0, be0)
  ln_kernel<<<TOK, 128, 0, stream>>>(dec_output, g0, be0, x_f32, x_bf, D);
  // 3) bf16 conversions (activations row-major; weights transposed to [N,K])
  cvt(enc_out_mem, enc_bf, MEMROWS * D);
  cvt(tgt_emb_mem, emb_bf, MEMROWS * D);
  cvtT(Wq, WqT, D, D);  cvtT(Wk, WkT, D, D);
  cvtT(Wv, WvT, D, D);  cvtT(Wo, WoT, D, D);
  cvtT(W1, W1T, D, FF); cvtT(W2, W2T, FF, D);
  cvtT(W3, W3T, D, FF); cvtT(W4, W4T, FF, D);

  // 4) Kp = enc_out_mem @ Wk + bk  (shared across all tokens selecting entry m)
  launch_gemm(enc_bf, WkT, bk, nullptr, nullptr, Kp_bf,
              MEMROWS, D, D, 0, stream);
  // 5) Vp = tgt_emb_mem @ Wv + bv
  launch_gemm(emb_bf, WvT, bv, nullptr, nullptr, Vp_bf,
              MEMROWS, D, D, 0, stream);
  // 6) q = x @ Wq + bq  (f32 out; emb region is dead now)
  launch_gemm(x_bf, WqT, bq, nullptr, q_f32, nullptr,
              TOK, D, D, 0, stream);
  // 7) per-token attention -> ctx (bf16)
  attention_kernel<<<TOK, 128, 0, stream>>>(q_f32, Kp_bf, Vp_bf, tgt_mask_mem,
                                            samples, ctx_bf);
  // 8) st = ctx @ Wo + bo + x
  launch_gemm(ctx_bf, WoT, bo, x_f32, st_f32, st_bf,
              TOK, D, D, 0, stream);
  // 9) h = relu(st @ W1 + bf1)
  launch_gemm(st_bf, W1T, bf1, nullptr, nullptr, h_bf,
              TOK, FF, D, 1, stream);
  // 10) st2 = st + h @ W2 + bf2
  launch_gemm(h_bf, W2T, bf2, st_f32, st2, nullptr,
              TOK, D, FF, 0, stream);
  // 11) x2 = x + (valid ? LN(st2) : 0)
  ln_scatter_kernel<<<TOK, 128, 0, stream>>>(st2, g1, be1, x_f32, samples,
                                             x2, x2_bf, D);
  // 12) h2 = relu(x2 @ W3 + bf3)
  launch_gemm(x2_bf, W3T, bf3, nullptr, nullptr, h2_bf,
              TOK, FF, D, 1, stream);
  // 13) out = x2 + h2 @ W4 + bf4
  launch_gemm(h2_bf, W4T, bf4, x2, out, nullptr,
              TOK, D, FF, 0, stream);
}
